// Model_3899830305219
// MI455X (gfx1250) — compile-verified, matrix operand order checked by reading, not run on hardware
//
#include <hip/hip_runtime.h>
#include <hip/hip_bf16.h>
#include <stdint.h>

// ---------------------------------------------------------------------------
// out[s] = 0.5 * sum_{i,j,n,a,b} W[a,b,n] A[i,j,n] v[i,a,s] v[j,b,s]
// Dominant work: t[i,n,(b,s)] = sum_j A[i,j,n] v[j,b,s]  (4x GEMM 4096x4096x192)
// bf16 WMMA (v_wmma_f32_16x16x32_bf16), f32 accumulation. A streamed from HBM
// once (256MB ~ 11us @ 23.3TB/s), converted f32->bf16 in regs, staged in LDS.
// B (pre-swizzled bf16 v) is copied to LDS per K-step with
// global_load_async_to_lds_b128 (ASYNCcnt) and double-buffered, so WMMAs only
// ever wait on DScnt, never on the HBM A stream. Epilogue contracts against
// precomputed vW[i,n,col] and reduces via LDS + global float atomics.
// ---------------------------------------------------------------------------

typedef __attribute__((ext_vector_type(16))) __bf16 v16bf;
typedef __attribute__((ext_vector_type(8)))  float  v8f;

#define NSITES   4096
#define NTYPES   3
#define NORDER   4
#define NBATCH   64
#define NCOLS    192                 // NTYPES * NBATCH
#define KTILE    32                  // K per wmma_16x16x32
#define NKT      (NSITES / KTILE)    // 128 k-steps
#define NCT      (NCOLS / 16)        // 12 column tiles
#define FRAG_ELEMS 512               // 32 lanes * 16 bf16 per 32x16 B fragment
#define BTILE_ELEMS (NCT * FRAG_ELEMS)   // 6144 bf16 = 12KB per k-step

// --- prep 1: pre-swizzle v (f32 [j,b,s]) into bf16 B fragments -------------
__global__ __launch_bounds__(256) void prep_b_kernel(const float* __restrict__ v,
                                                     __bf16* __restrict__ Bsw) {
    int idx = blockIdx.x * 256 + threadIdx.x;        // 0 .. 4096*192-1
    int j   = idx / NCOLS;
    int col = idx % NCOLS;
    int b = col >> 6, s = col & 63;
    float val = v[((size_t)j * NTYPES + b) * NBATCH + s];
    int kt = j >> 5, k = j & 31;
    int ct = col >> 4, ncol = col & 15;
    int ln   = ncol + ((k >> 3) & 1) * 16;
    int slot = (k & 7) + ((k >> 4) & 1) * 8;
    Bsw[(((size_t)kt * NCT + ct) * 32 + ln) * 16 + slot] = (__bf16)val;
}

// --- prep 2: vW[i,n,col] = sum_a W[a,b,n] * v[i,a,s],  col = b*64+s --------
__global__ __launch_bounds__(256) void prep_vw_kernel(const float* __restrict__ W,
                                                      const float* __restrict__ v,
                                                      float* __restrict__ vW) {
    size_t idx = (size_t)blockIdx.x * 256 + threadIdx.x;  // i*768 + n*192 + col
    int col = (int)(idx % NCOLS);
    size_t t2 = idx / NCOLS;
    int n = (int)(t2 & 3);
    int i = (int)(t2 >> 2);
    int b = col >> 6, s = col & 63;
    float acc = 0.0f;
#pragma unroll
    for (int a = 0; a < NTYPES; ++a)
        acc += W[(a * NTYPES + b) * NORDER + n] *
               v[((size_t)i * NTYPES + a) * NBATCH + s];
    vW[idx] = acc;
}

// --- prep 3: zero the 64-float output --------------------------------------
__global__ void zero_out_kernel(float* out) {
    if (threadIdx.x < NBATCH) out[threadIdx.x] = 0.0f;
}

// --- main: fused GEMM (bf16 WMMA) + energy reduction -----------------------
// 256 blocks x 256 threads (8 wave32 waves). Block owns 16 rows of i.
// wave w: n = w & 3, column-tile group = (w >> 2) * 6 .. +5  -> 6 C tiles.
__global__ __launch_bounds__(256) void energy_kernel(const float*  __restrict__ A,
                                                     const float*  __restrict__ vW,
                                                     const __bf16* __restrict__ Bsw,
                                                     float* __restrict__ out) {
    __shared__ __bf16 Alds[2][NORDER * 32 * 16];   // 2 x 4KB, [n][lane][slot]
    __shared__ __bf16 Blds[2][BTILE_ELEMS];        // 2 x 12KB, [ct][lane][slot]
    __shared__ float  outs[NBATCH];

    const int tid  = threadIdx.x;
    const int wave = tid >> 5;
    const int lane = tid & 31;
    const int n    = wave & 3;
    const int wc   = wave >> 2;
    const int i0   = blockIdx.x * 16;

    if (tid < NBATCH) outs[tid] = 0.0f;

    v8f acc[6] = {};

    const float4* Ag = reinterpret_cast<const float4*>(A);
    const int ai = tid >> 4;           // 0..15 : row in tile
    const int aj = (tid & 15) * 2;     // 0,2,..,30 : first of 2 j cells
    float4 r0, r1;

    auto loadA = [&](int kt) {
        size_t base = (size_t)(i0 + ai) * NSITES + (size_t)kt * KTILE + aj;
        r0 = Ag[base];
        r1 = Ag[base + 1];
    };
    auto store1 = [&](int buf, const float4& r, int j) {
        int ln   = ai + ((j >> 3) & 1) * 16;
        int slot = (j & 7) + ((j >> 4) & 1) * 8;
        Alds[buf][(0 * 32 + ln) * 16 + slot] = (__bf16)r.x;
        Alds[buf][(1 * 32 + ln) * 16 + slot] = (__bf16)r.y;
        Alds[buf][(2 * 32 + ln) * 16 + slot] = (__bf16)r.z;
        Alds[buf][(3 * 32 + ln) * 16 + slot] = (__bf16)r.w;
    };
    auto storeA = [&](int buf) { store1(buf, r0, aj); store1(buf, r1, aj + 1); };

    // Async copy of one 12KB B tile into LDS: 3 chunks of 4KB; each lane moves
    // 16B per instruction (global_load_async_to_lds_b128, tracked by ASYNCcnt).
    auto asyncB = [&](int kt, int buf) {
        const char* g = (const char*)(Bsw + (size_t)kt * BTILE_ELEMS);
        unsigned lbase = (unsigned)(uintptr_t)(&Blds[buf][0]);
#pragma unroll
        for (int c = 0; c < 3; ++c) {
            unsigned off = (unsigned)(c * 4096 + tid * 16);
            unsigned laddr = lbase + off;
            const char* gaddr = g + off;
            asm volatile("global_load_async_to_lds_b128 %0, %1, off"
                         :: "v"(laddr), "v"(gaddr) : "memory");
        }
    };
    auto waitAsync = [&]() {
        asm volatile("s_wait_asynccnt 0x0" ::: "memory");
    };

    // ---- prologue: fill buffer 0 ------------------------------------------
    loadA(0);
    asyncB(0, 0);
    storeA(0);
    loadA(1);
    waitAsync();
    __syncthreads();

    // ---- main K loop: one barrier per iteration, double-buffered ----------
    for (int kt = 0; kt < NKT; ++kt) {
        const int cur = kt & 1, nxt = cur ^ 1;
        if (kt + 1 < NKT) {
            asyncB(kt + 1, nxt);           // ASYNCcnt copy overlaps WMMAs
            storeA(nxt);                   // stage A tile kt+1 (regs -> LDS)
            if (kt + 2 < NKT) loadA(kt + 2);   // keep HBM stream ahead
            if (kt + 4 < NKT)              // near-scope prefetch of A stream
                __builtin_prefetch(&Ag[(size_t)(i0 + ai) * NSITES +
                                       (size_t)(kt + 4) * KTILE + aj], 0, 3);
        }
        // operands for this K-step (all from LDS: DScnt only, no HBM waits)
        v16bf af = *reinterpret_cast<const v16bf*>(&Alds[cur][(n * 32 + lane) * 16]);
        v16bf bf[6];
#pragma unroll
        for (int q = 0; q < 6; ++q)
            bf[q] = *reinterpret_cast<const v16bf*>(
                &Blds[cur][((wc * 6 + q) * 32 + lane) * 16]);
#pragma unroll
        for (int q = 0; q < 6; ++q)
            acc[q] = __builtin_amdgcn_wmma_f32_16x16x32_bf16(
                false, af, false, bf[q], (short)0, acc[q], false, false);
        if (kt + 1 < NKT) waitAsync();     // next B tile landed in LDS
        __syncthreads();                   // publish Alds[nxt] / Blds[nxt]
    }

    // ---- epilogue: t[i,n,col] (in acc) . vW[i,n,col] -> out[s] ------------
    // C/D layout: VGPR r, lanes 0-15 -> M=r, N=lane; lanes 16-31 -> M=r+8.
    const int colLocal = lane & 15;
    const int mhalf    = (lane >> 4) * 8;
#pragma unroll
    for (int q = 0; q < 6; ++q) {
        int col = (wc * 6 + q) * 16 + colLocal;
        float partial = 0.0f;
#pragma unroll
        for (int r = 0; r < 8; ++r) {
            int i = i0 + mhalf + r;
            partial += acc[q][r] * vW[((size_t)i * NORDER + n) * NCOLS + col];
        }
        atomicAdd(&outs[col & 63], partial);   // ds_add_f32
    }
    __syncthreads();
    if (tid < NBATCH) atomicAdd(&out[tid], 0.5f * outs[tid]);
}

extern "C" void kernel_launch(void* const* d_in, const int* in_sizes, int n_in,
                              void* d_out, int out_size, void* d_ws, size_t ws_size,
                              hipStream_t stream) {
    const float* W = (const float*)d_in[0];   // [3,3,4]
    const float* A = (const float*)d_in[1];   // [4096,4096,4]
    const float* v = (const float*)d_in[2];   // [4096,3,64]
    float* out = (float*)d_out;               // [64]

    // workspace: bf16 pre-swizzled v (1.5MB), then f32 vW (12.6MB)
    __bf16* Bsw = (__bf16*)d_ws;
    float*  vW  = (float*)((char*)d_ws + (size_t)NKT * BTILE_ELEMS * sizeof(__bf16));

    prep_b_kernel<<<(NSITES * NCOLS) / 256, 256, 0, stream>>>(v, Bsw);
    prep_vw_kernel<<<(NSITES * NORDER * NCOLS) / 256, 256, 0, stream>>>(W, v, vW);
    zero_out_kernel<<<1, 64, 0, stream>>>(out);
    energy_kernel<<<NSITES / 16, 256, 0, stream>>>(A, vW, Bsw, out);
}
